// HybridAttention_59425167508019
// MI455X (gfx1250) — matrix operand order repaired and checked by reading, not compiled, not run on hardware
//
#include <hip/hip_runtime.h>
#include <hip/hip_bf16.h>

typedef _Float16 h16 __attribute__((ext_vector_type(16)));
typedef _Float16 h8  __attribute__((ext_vector_type(8)));
typedef float    f8  __attribute__((ext_vector_type(8)));

#define BB 4
#define LL 2048
#define HH 4    // heads per branch
#define DD 64
#define NH 8    // total output heads (AGF 0..3, softmax 4..7)

// ---------------- WMMA helpers (gfx1250, wave32) ----------------

__device__ __forceinline__ f8 wmma_f16(h16 a, h16 b, f8 c) {
  return __builtin_amdgcn_wmma_f32_16x16x32_f16(false, a, false, b, (short)0, c,
                                                false, false);
}

// A fragment (16x32 f16): lane -> M = lane&15; K runs {0..7|8..15}+{16..23|24..31}
__device__ __forceinline__ h16 fragA(const _Float16* row, int kbase, int lane) {
  int o = (lane & 16) ? 8 : 0;
  h8 lo = *(const h8*)(row + kbase + o);
  h8 hi = *(const h8*)(row + kbase + 16 + o);
  h16 a;
#pragma unroll
  for (int i = 0; i < 8; ++i) { a[i] = lo[i]; a[8 + i] = hi[i]; }
  return a;
}

// B fragment (32x16 f16): lane -> N = lane&15; K runs {0..7|8..15}+{16..23|24..31}
__device__ __forceinline__ h16 fragB(const _Float16* row, int kbase, int lane) {
  int o = (lane & 16) ? 8 : 0;
  h8 lo = *(const h8*)(row + kbase + o);
  h8 hi = *(const h8*)(row + kbase + 16 + o);
  h16 b;
#pragma unroll
  for (int i = 0; i < 8; ++i) { b[i] = lo[i]; b[8 + i] = hi[i]; }
  return b;
}

// ---------------- K0: zero the loss outputs ----------------

__global__ void zero_loss_kernel(float* __restrict__ loss) {
  if (threadIdx.x < BB) loss[threadIdx.x] = 0.f;
}

// ---------------- K1: Uh softmax + sigmoid + Jacobi filter ----------------
// one wave per (b,h,l) row; writes Ug (f16, [BH][L][64]) and UhT (f16, [BH][64][L])

__global__ void prep_u_kernel(const float* __restrict__ U,
                              const float* __restrict__ Sg,
                              const float* __restrict__ gam,
                              _Float16* __restrict__ Ug,
                              _Float16* __restrict__ UhT) {
  int wave = threadIdx.x >> 5, lane = threadIdx.x & 31;
  int row = blockIdx.x * 8 + wave;   // over B*H*L
  int l  = row & (LL - 1);
  int bh = row >> 11;
  int b = bh >> 2, h = bh & 3;
  size_t ibase = (((size_t)(b * LL + l)) * HH + h) * DD + lane * 2;
  float2 u = *(const float2*)(U + ibase);
  float2 s = *(const float2*)(Sg + ibase);
  float m = fmaxf(u.x, u.y);
#pragma unroll
  for (int o = 16; o; o >>= 1) m = fmaxf(m, __shfl_xor(m, o, 32));
  float e0 = __expf(u.x - m), e1 = __expf(u.y - m);
  float sum = e0 + e1;
#pragma unroll
  for (int o = 16; o; o >>= 1) sum += __shfl_xor(sum, o, 32);
  float inv = 1.f / sum;
  float uh0 = e0 * inv, uh1 = e1 * inv;
  float g0 = gam[0], g1 = gam[1], g2 = gam[2], g3 = gam[3];
  // Jacobi(a=b=1): P0=1, P1=2x, P2=1.875x*P1-0.75, P3=(336/180)x*P2-(144/180)P1
  auto gf = [&](float x) {
    float sg = 1.f / (1.f + __expf(-x));
    float p1 = 2.f * sg;
    float p2 = 1.875f * sg * p1 - 0.75f;
    float p3 = (336.f / 180.f) * sg * p2 - (144.f / 180.f) * p1;
    return g0 + g1 * p1 + g2 * p2 + g3 * p3;
  };
  size_t ob = ((size_t)bh * LL + l) * DD + lane * 2;
  Ug[ob]     = (_Float16)(uh0 * gf(s.x));
  Ug[ob + 1] = (_Float16)(uh1 * gf(s.y));
  size_t tb = ((size_t)bh * DD + lane * 2) * LL + l;
  UhT[tb]      = (_Float16)uh0;
  UhT[tb + LL] = (_Float16)uh1;
}

// ---------------- K2: Vt column-softmax over L + vals transpose ----------------
// one block per (b,h,p); writes Vt and ValsT (f16, [BH][64][L])

__global__ void prep_v_kernel(const float* __restrict__ V,
                              const float* __restrict__ Vals,
                              _Float16* __restrict__ Vt,
                              _Float16* __restrict__ ValsT) {
  __shared__ float redm[8];
  __shared__ float reds[8];
  int p  = blockIdx.x & 63;
  int bh = blockIdx.x >> 6;
  int b = bh >> 2, h = bh & 3;
  int t = threadIdx.x, wave = t >> 5, lane = t & 31;
  float v[8];
#pragma unroll
  for (int j = 0; j < 8; ++j) {
    int l = t + j * 256;
    v[j] = V[(((size_t)(b * LL + l)) * HH + h) * DD + p];
  }
  float m = v[0];
#pragma unroll
  for (int j = 1; j < 8; ++j) m = fmaxf(m, v[j]);
#pragma unroll
  for (int o = 16; o; o >>= 1) m = fmaxf(m, __shfl_xor(m, o, 32));
  if (lane == 0) redm[wave] = m;
  __syncthreads();
  float mm = redm[0];
#pragma unroll
  for (int i = 1; i < 8; ++i) mm = fmaxf(mm, redm[i]);
  float e[8], sum = 0.f;
#pragma unroll
  for (int j = 0; j < 8; ++j) { e[j] = __expf(v[j] - mm); sum += e[j]; }
#pragma unroll
  for (int o = 16; o; o >>= 1) sum += __shfl_xor(sum, o, 32);
  if (lane == 0) reds[wave] = sum;
  __syncthreads();
  float ss = 0.f;
#pragma unroll
  for (int i = 0; i < 8; ++i) ss += reds[i];
  float inv = 1.f / ss;
#pragma unroll
  for (int j = 0; j < 8; ++j) {
    int l = t + j * 256;
    size_t o2 = ((size_t)bh * DD + p) * LL + l;
    Vt[o2]    = (_Float16)(e[j] * inv);
    ValsT[o2] = (_Float16)Vals[(((size_t)(b * LL + l)) * HH + h) * DD + p];
  }
}

// ---------------- K3: kv[b,h] = Vt(64xL) @ vals(Lx64) ----------------
// one block per (b,h); 16 waves, one 16x16 WMMA tile each, K=2048

__global__ void kv_gemm_kernel(const _Float16* __restrict__ Vt,
                               const _Float16* __restrict__ ValsT,
                               float* __restrict__ KV) {
  int bh = blockIdx.x;
  int wave = threadIdx.x >> 5, lane = threadIdx.x & 31;
  int mt = wave >> 2, nt = wave & 3;
  const _Float16* arow = Vt    + ((size_t)bh * DD + mt * 16 + (lane & 15)) * LL;
  const _Float16* brow = ValsT + ((size_t)bh * DD + nt * 16 + (lane & 15)) * LL;
  f8 acc = {};
  for (int k = 0; k < LL; k += 32)
    acc = wmma_f16(fragA(arow, k, lane), fragB(brow, k, lane), acc);
  int Mo = (lane & 16) ? 8 : 0, N = lane & 15;
#pragma unroll
  for (int r = 0; r < 8; ++r)
    KV[((size_t)bh * DD + mt * 16 + Mo + r) * DD + nt * 16 + N] = acc[r];
}

// ---------------- K4: out_agf = Ug(Lx64) @ kv(64x64), write heads 0..3 ----------------

__global__ void agf_out_kernel(const _Float16* __restrict__ Ug,
                               const float* __restrict__ KV,
                               float* __restrict__ out) {
  __shared__ __attribute__((aligned(32))) _Float16 kvT[DD][DD];
  int bh = blockIdx.y, b = bh >> 2, h = bh & 3;
  for (int i = threadIdx.x; i < DD * DD; i += 256) {
    int p = i >> 6, e = i & 63;
    kvT[e][p] = (_Float16)KV[(size_t)bh * DD * DD + i];
  }
  __syncthreads();
  int wave = threadIdx.x >> 5, lane = threadIdx.x & 31;
  int Mbase = blockIdx.x * 128 + wave * 16;
  const _Float16* arow = Ug + ((size_t)bh * LL + Mbase + (lane & 15)) * DD;
  f8 acc[4] = {};
#pragma unroll
  for (int k = 0; k < DD; k += 32) {
    h16 a = fragA(arow, k, lane);
#pragma unroll
    for (int nt = 0; nt < 4; ++nt)
      acc[nt] = wmma_f16(a, fragB(&kvT[nt * 16 + (lane & 15)][0], k, lane), acc[nt]);
  }
  int Mo = (lane & 16) ? 8 : 0, N = lane & 15;
#pragma unroll
  for (int nt = 0; nt < 4; ++nt)
#pragma unroll
    for (int r = 0; r < 8; ++r) {
      int l = Mbase + Mo + r;
      out[(((size_t)b * LL + l) * NH + h) * DD + nt * 16 + N] = acc[nt][r];
    }
}

// ---------------- K5: ortho loss: |M M^T - I| reduction (M = UhT or Vt) ----------------

__global__ void sym_gemm_kernel(const _Float16* __restrict__ UhT,
                                const _Float16* __restrict__ Vt,
                                float* __restrict__ loss) {
  __shared__ float red[16];
  int bh = blockIdx.x, b = bh >> 2;
  const _Float16* M = blockIdx.y ? Vt : UhT;
  int wave = threadIdx.x >> 5, lane = threadIdx.x & 31;
  int mt = wave >> 2, nt = wave & 3;
  const _Float16* arow = M + ((size_t)bh * DD + mt * 16 + (lane & 15)) * LL;
  const _Float16* brow = M + ((size_t)bh * DD + nt * 16 + (lane & 15)) * LL;
  f8 acc = {};
  for (int k = 0; k < LL; k += 32)
    acc = wmma_f16(fragA(arow, k, lane), fragB(brow, k, lane), acc);
  int gm0 = mt * 16 + ((lane & 16) ? 8 : 0), gn = nt * 16 + (lane & 15);
  float part = 0.f;
#pragma unroll
  for (int r = 0; r < 8; ++r)
    part += fabsf(acc[r] - ((gm0 + r == gn) ? 1.f : 0.f));
#pragma unroll
  for (int o = 16; o; o >>= 1) part += __shfl_xor(part, o, 32);
  if (lane == 0) red[wave] = part;
  __syncthreads();
  if (threadIdx.x == 0) {
    float tot = 0.f;
#pragma unroll
    for (int i = 0; i < 16; ++i) tot += red[i];
    atomicAdd(loss + b, tot * (1.f / 16384.f));
  }
}

// ---------------- K6: flash attention for softmax heads (write heads 4..7) ----------------

__global__ void flash_attn_kernel(const float* __restrict__ Q,
                                  const float* __restrict__ Km,
                                  const float* __restrict__ Vm,
                                  float* __restrict__ out) {
  __shared__ __attribute__((aligned(32))) _Float16 q_s[128][DD];
  __shared__ __attribute__((aligned(32))) _Float16 k_s[32][DD];
  __shared__ __attribute__((aligned(32))) _Float16 vt_s[DD][32];
  __shared__ __attribute__((aligned(32))) _Float16 p_s[8][16][32];
  int bh = blockIdx.y, b = bh >> 2, h = bh & 3;
  int qbase = blockIdx.x * 128;
  int t = threadIdx.x, wave = t >> 5, lane = t & 31;
  for (int i = t; i < 128 * DD; i += 256) {
    int r = i >> 6, e = i & 63;
    q_s[r][e] = (_Float16)Q[(((size_t)(b * LL + qbase + r)) * HH + h) * DD + e];
  }
  __syncthreads();
  h16 aq0 = fragA(&q_s[wave * 16 + (lane & 15)][0], 0, lane);
  h16 aq1 = fragA(&q_s[wave * 16 + (lane & 15)][0], 32, lane);
  f8 acc[4] = {};
  float rm[8], rs[8];
#pragma unroll
  for (int r = 0; r < 8; ++r) { rm[r] = -INFINITY; rs[r] = 0.f; }
  for (int s0 = 0; s0 < LL; s0 += 32) {
    __syncthreads();
    for (int i = t; i < 32 * DD; i += 256) {
      int r = i >> 6, e = i & 63;
      size_t gi = (((size_t)(b * LL + s0 + r)) * HH + h) * DD + e;
      k_s[r][e]  = (_Float16)Km[gi];
      vt_s[e][r] = (_Float16)Vm[gi];
    }
    __syncthreads();
    f8 sa0 = {}, sa1 = {};
#pragma unroll
    for (int ks = 0; ks < DD; ks += 32) {
      h16 a = ks ? aq1 : aq0;
      sa0 = wmma_f16(a, fragB(&k_s[lane & 15][0], ks, lane), sa0);
      sa1 = wmma_f16(a, fragB(&k_s[16 + (lane & 15)][0], ks, lane), sa1);
    }
    int Mo = (lane & 16) ? 8 : 0;
#pragma unroll
    for (int r = 0; r < 8; ++r) {
      float x0 = sa0[r] * 0.125f, x1 = sa1[r] * 0.125f;
      float cm = fmaxf(x0, x1);
#pragma unroll
      for (int o = 8; o; o >>= 1) cm = fmaxf(cm, __shfl_xor(cm, o, 32));
      float mn = fmaxf(rm[r], cm);
      float sc = __expf(rm[r] - mn);
      float p0 = __expf(x0 - mn), p1 = __expf(x1 - mn);
      float ps = p0 + p1;
#pragma unroll
      for (int o = 8; o; o >>= 1) ps += __shfl_xor(ps, o, 32);
      rs[r] = rs[r] * sc + ps;
      rm[r] = mn;
#pragma unroll
      for (int nt = 0; nt < 4; ++nt) acc[nt][r] *= sc;
      p_s[wave][Mo + r][lane & 15]      = (_Float16)p0;
      p_s[wave][Mo + r][16 + (lane & 15)] = (_Float16)p1;
    }
    h16 ap = fragA(&p_s[wave][lane & 15][0], 0, lane);
#pragma unroll
    for (int nt = 0; nt < 4; ++nt)
      acc[nt] = wmma_f16(ap, fragB(&vt_s[nt * 16 + (lane & 15)][0], 0, lane), acc[nt]);
  }
  float invs[8];
  int Mo = (lane & 16) ? 8 : 0, N = lane & 15;
#pragma unroll
  for (int r = 0; r < 8; ++r) invs[r] = 1.f / rs[r];
#pragma unroll
  for (int nt = 0; nt < 4; ++nt)
#pragma unroll
    for (int r = 0; r < 8; ++r) {
      int l = qbase + wave * 16 + Mo + r;
      out[(((size_t)b * LL + l) * NH + (HH + h)) * DD + nt * 16 + N] =
          acc[nt][r] * invs[r];
    }
}

// ---------------- launch ----------------

extern "C" void kernel_launch(void* const* d_in, const int* in_sizes, int n_in,
                              void* d_out, int out_size, void* d_ws, size_t ws_size,
                              hipStream_t stream) {
  const float* U  = (const float*)d_in[0];
  const float* Sg = (const float*)d_in[1];
  const float* V  = (const float*)d_in[2];
  const float* Vl = (const float*)d_in[3];
  const float* Qs = (const float*)d_in[4];
  const float* Ks = (const float*)d_in[5];
  const float* Vs = (const float*)d_in[6];
  const float* gm = (const float*)d_in[7];
  float* out  = (float*)d_out;
  float* loss = out + (size_t)BB * LL * NH * DD;

  char* ws = (char*)d_ws;
  _Float16* Ug    = (_Float16*)(ws);
  _Float16* UhT   = (_Float16*)(ws + ((size_t)4 << 20));
  _Float16* Vt    = (_Float16*)(ws + ((size_t)8 << 20));
  _Float16* ValsT = (_Float16*)(ws + ((size_t)12 << 20));
  float*    KV    = (float*)   (ws + ((size_t)16 << 20));

  hipLaunchKernelGGL(zero_loss_kernel, dim3(1), dim3(32), 0, stream, loss);
  hipLaunchKernelGGL(prep_u_kernel, dim3(4096), dim3(256), 0, stream, U, Sg, gm, Ug, UhT);
  hipLaunchKernelGGL(prep_v_kernel, dim3(1024), dim3(256), 0, stream, V, Vl, Vt, ValsT);
  hipLaunchKernelGGL(kv_gemm_kernel, dim3(16), dim3(512), 0, stream, Vt, ValsT, KV);
  hipLaunchKernelGGL(agf_out_kernel, dim3(16, 16), dim3(256), 0, stream, Ug, KV, out);
  hipLaunchKernelGGL(sym_gemm_kernel, dim3(16, 2), dim3(512), 0, stream, UhT, Vt, loss);
  hipLaunchKernelGGL(flash_attn_kernel, dim3(16, 16), dim3(256), 0, stream, Qs, Ks, Vs, out);
}
